// VectorQuantizer2_58978490909118
// MI455X (gfx1250) — matrix operand order repaired and checked by reading, not compiled, not run on hardware
//
#include <hip/hip_runtime.h>
#include <hip/hip_bf16.h>
#include <math.h>

typedef __attribute__((ext_vector_type(16))) _Float16 v16h;
typedef __attribute__((ext_vector_type(8)))  float    v8f;
typedef unsigned int u32x4 __attribute__((ext_vector_type(4)));
typedef int          i32x8 __attribute__((ext_vector_type(8)));
typedef int          i32x4 __attribute__((ext_vector_type(4)));

#define CDIM 64
#define NBATCH 96
#define NFELEM (96*64*16*16)
#define VOCAB 4096

// ---------------------------------------------------------------------------
// helpers
// ---------------------------------------------------------------------------
__device__ __forceinline__ float silu_f(float x) {
  return x * (1.0f / (1.0f + __expf(-x)));
}

// Unified conv / lhs-dilated(transposed) conv source fetch (compile-time shape).
// normal conv:      OS=s, PD=p,      LD=1
// transposed conv:  OS=1, PD=k-1-p,  LD=s
template<int KSZ, int OS, int PD, int LD>
__device__ __forceinline__ float conv_src(const float* __restrict__ in, int b, int oy, int ox,
                                          int K, int IH, int IW) {
  constexpr int KK2 = KSZ * KSZ;
  int ci = K / KK2;
  int r  = K - ci * KK2;
  int ky = r / KSZ;
  int kx = r - ky * KSZ;
  int ty = oy * OS - PD + ky;
  int tx = ox * OS - PD + kx;
  if (ty < 0 || tx < 0) return 0.0f;
  int iy, ix;
  if constexpr (LD == 1) {
    iy = ty; ix = tx;
  } else {
    iy = ty / LD; ix = tx / LD;
    if ((ty - iy * LD) | (tx - ix * LD)) return 0.0f;
  }
  if (iy >= IH || ix >= IW) return 0.0f;
  return in[((b * CDIM + ci) * IH + iy) * IW + ix];
}

// ---------------------------------------------------------------------------
// TDM: DMA the 32(K) x 64(Cout) f32 weight tile for one K-step into LDS.
// D# built per CDNA5 ISA (2D tensor, groups 2/3 zero).
//   rows: n = 0..63 (stride Ktot f32), cols: 32 contiguous f32 at global_addr.
//   LDS result: wtile[n*32 + kk], compact row-major.
// 6-arg builtin form (clang-23 / therock-10.0 headers).
// ---------------------------------------------------------------------------
__device__ __forceinline__ void tdm_load_wtile(const float* gsrc, unsigned lds_addr, int Ktot) {
  unsigned long long ga = (unsigned long long)(size_t)gsrc;
  u32x4 g0;
  g0[0] = 1u;                                                // count=1 (valid user D#)
  g0[1] = lds_addr;                                          // LDS byte address
  g0[2] = (unsigned)ga;                                      // global_addr[31:0]
  g0[3] = (unsigned)((ga >> 32) & 0x1FFFFFFu) | (2u << 30);  // global_addr[56:32] | type=2
  i32x8 g1;
  g1[0] = 2 << 16;                                           // data_size = 4 bytes
  g1[1] = (Ktot & 0xFFFF) << 16;                             // tensor_dim0[15:0]
  g1[2] = (int)(((unsigned)Ktot >> 16) | (64u << 16));       // tensor_dim0[31:16] | tensor_dim1=64
  g1[3] = 32 << 16;                                          // tile_dim0 = 32
  g1[4] = 64;                                                // tile_dim1 = 64 (tile_dim2 = 0)
  g1[5] = Ktot;                                              // tensor_dim0_stride[31:0]
  g1[6] = 0;
  g1[7] = 0;
  i32x4 z4 = {0, 0, 0, 0};
  i32x8 z8 = {0, 0, 0, 0, 0, 0, 0, 0};
  __builtin_amdgcn_tensor_load_to_lds(g0, g1, z4, z4, z8, 0);
}

// ---------------------------------------------------------------------------
// init: f_rest = f ; f_hat(d_out) = 0 ; loss slot = 0
// ---------------------------------------------------------------------------
__global__ void vq_init_kernel(const float* __restrict__ f, float* __restrict__ frest,
                               float* __restrict__ out) {
  int i = blockIdx.x * blockDim.x + threadIdx.x;
  if (i <= NFELEM) {
    out[i] = 0.0f;
    if (i < NFELEM) frest[i] = f[i];
  }
}

// ---------------------------------------------------------------------------
// codebook squared norms
// ---------------------------------------------------------------------------
__global__ void vq_wsq_kernel(const float* __restrict__ cb, float* __restrict__ wsq) {
  int n = blockIdx.x * blockDim.x + threadIdx.x;
  if (n < VOCAB) {
    float s = 0.0f;
    #pragma unroll
    for (int c = 0; c < CDIM; ++c) { float v = cb[n * CDIM + c]; s += v * v; }
    wsq[n] = s;
  }
}

// ---------------------------------------------------------------------------
// WMMA implicit-GEMM conv (CI=CO=64), shapes as template constants.
// Weights staged per-K-step into LDS by TDM (wave 0), read back as ds_load_b64.
// mode 0: out = silu(conv+bias)
// mode 1 (Phi): v = 0.5*hin + 0.5*(conv+bias); fhat += v; frest -= v
// One wave computes a 16-row (token) x 64-col (out-channel) tile.
// ---------------------------------------------------------------------------
template<int KSZ, int OS, int PD, int LD>
__global__ void conv_wmma_kernel(const float* __restrict__ in, const float* __restrict__ w,
                                 const float* __restrict__ bias, float* __restrict__ out,
                                 int IH, int OH, int mode, const float* __restrict__ hin,
                                 float* __restrict__ fhat, float* __restrict__ frest) {
  constexpr int Ktot = CDIM * KSZ * KSZ;   // always a multiple of 32
  const int IW = IH, OW = OH;

  __shared__ float wtile[CDIM * 32];       // 8 KB: [n][kk] for one K-step
  const unsigned wlds = (unsigned)(size_t)&wtile[0];

  const int lane = threadIdx.x & 31;
  const int wv   = threadIdx.x >> 5;
  const int M = NBATCH * OH * OW;
  const int Mtiles = (M + 15) >> 4;
  const int tile  = blockIdx.x * (blockDim.x >> 5) + wv;
  const int tileC = (tile < Mtiles) ? tile : (Mtiles - 1);   // clamp: keep barriers uniform

  const int g  = lane >> 4;       // half-wave group
  const int ln = lane & 15;       // column within tile / row for A

  int mA = tileC * 16 + ln; if (mA >= M) mA = M - 1;
  const int ox = mA % OW; int t0 = mA / OW;
  const int oy = t0 % OH; const int b = t0 / OH;

  v8f acc[4];
  #pragma unroll
  for (int nt = 0; nt < 4; ++nt)
    #pragma unroll
    for (int i = 0; i < 8; ++i) acc[nt][i] = 0.0f;

  for (int kt = 0; kt < Ktot; kt += 32) {
    __syncthreads();                       // previous K-step's LDS reads complete
    if (wv == 0) {
      tdm_load_wtile(w + kt, wlds, Ktot);  // one TDM DMA per block per K-step
      __builtin_amdgcn_s_wait_tensorcnt(0);
    }
    __syncthreads();                       // tile visible to all waves

    // A fragment: 16x32 f16, ISA layout (lanes 0-15: K 0-7 & 16-23; 16-31: K 8-15 & 24-31)
    v16h afrag;
    #pragma unroll
    for (int j = 0; j < 8; ++j) {
      int K0 = kt + ((j & 3) << 1) + ((j >> 2) << 4) + (g << 3);
      afrag[2 * j]     = (_Float16)conv_src<KSZ, OS, PD, LD>(in, b, oy, ox, K0,     IH, IW);
      afrag[2 * j + 1] = (_Float16)conv_src<KSZ, OS, PD, LD>(in, b, oy, ox, K0 + 1, IH, IW);
    }
    #pragma unroll
    for (int nt = 0; nt < 4; ++nt) {
      int n = nt * 16 + ln;
      const float2* wp2 = (const float2*)&wtile[n * 32 + (g << 3)];
      v16h bfrag;  // B: lane holds col n; VGPR j holds K = g*8 + 2(j&3) + 16(j>>2) + {0,1}
      #pragma unroll
      for (int j = 0; j < 8; ++j) {
        float2 wvv = wp2[(j & 3) + ((j >> 2) << 3)];
        bfrag[2 * j]     = (_Float16)wvv.x;
        bfrag[2 * j + 1] = (_Float16)wvv.y;
      }
      acc[nt] = __builtin_amdgcn_wmma_f32_16x16x32_f16(
          false, afrag, false, bfrag, (short)0, acc[nt], false, false);
    }
  }

  if (tile >= Mtiles) return;  // after all barriers

  // epilogue: C layout — VGPR i holds row i+8g, column ln (+16*nt)
  #pragma unroll
  for (int nt = 0; nt < 4; ++nt) {
    int co = nt * 16 + ln;
    float bv = bias[co];
    #pragma unroll
    for (int i = 0; i < 8; ++i) {
      int m = tile * 16 + i + (g << 3);
      if (m >= M) continue;
      int ox2 = m % OW; int tt = m / OW;
      int oy2 = tt % OH; int b2 = tt / OH;
      int oidx = ((b2 * CDIM + co) * OH + oy2) * OW + ox2;
      float v = acc[nt][i] + bv;
      if (mode == 0) {
        out[oidx] = silu_f(v);
      } else {
        float x = 0.5f * hin[oidx] + 0.5f * v;
        fhat[oidx]  += x;
        frest[oidx] -= x;
      }
    }
  }
}

// ---------------------------------------------------------------------------
// WMMA codebook argmin: score = z.W - 0.5*||W||^2  (argmax score == argmin dist)
// One wave handles 16 tokens against all 4096 codes.
// ---------------------------------------------------------------------------
__global__ void argmin_wmma_kernel(const float* __restrict__ z, const float* __restrict__ cb,
                                   const float* __restrict__ wsq, int* __restrict__ idxOut,
                                   int pn) {
  const int lane = threadIdx.x & 31;
  const int tile = blockIdx.x * (blockDim.x >> 5) + (threadIdx.x >> 5);
  const int M = NBATCH * pn * pn;
  const int Mtiles = (M + 15) >> 4;
  if (tile >= Mtiles) return;

  const int g  = lane >> 4;
  const int ln = lane & 15;

  int mA = tile * 16 + ln; if (mA >= M) mA = M - 1;
  const int x = mA % pn; int t0 = mA / pn;
  const int y = t0 % pn; const int b = t0 / pn;

  // Preload both K-tiles of A (token features, NCHW gather + cvt to f16)
  v16h a[2];
  #pragma unroll
  for (int kt = 0; kt < 2; ++kt)
    #pragma unroll
    for (int j = 0; j < 8; ++j) {
      int K0 = kt * 32 + ((j & 3) << 1) + ((j >> 2) << 4) + (g << 3);
      a[kt][2 * j]     = (_Float16)z[((b * CDIM + K0)     * pn + y) * pn + x];
      a[kt][2 * j + 1] = (_Float16)z[((b * CDIM + K0 + 1) * pn + y) * pn + x];
    }

  float best[8]; int bidx[8];
  #pragma unroll
  for (int i = 0; i < 8; ++i) { best[i] = -1e30f; bidx[i] = 0; }

  for (int nt = 0; nt < VOCAB / 16; ++nt) {
    int n = nt * 16 + ln;
    float c0 = -0.5f * wsq[n];
    v8f acc;
    #pragma unroll
    for (int i = 0; i < 8; ++i) acc[i] = c0;
    #pragma unroll
    for (int kt = 0; kt < 2; ++kt) {
      const float* cp = cb + (size_t)n * CDIM + kt * 32 + (g << 3);
      v16h bfrag;
      #pragma unroll
      for (int j = 0; j < 8; ++j) {
        float2 wvv = *(const float2*)(cp + 2 * (j & 3) + ((j >> 2) << 4));
        bfrag[2 * j]     = (_Float16)wvv.x;
        bfrag[2 * j + 1] = (_Float16)wvv.y;
      }
      acc = __builtin_amdgcn_wmma_f32_16x16x32_f16(
          false, a[kt], false, bfrag, (short)0, acc, false, false);
    }
    #pragma unroll
    for (int i = 0; i < 8; ++i)
      if (acc[i] > best[i]) { best[i] = acc[i]; bidx[i] = n; }  // strict > keeps lowest idx
  }

  // reduce across the 16 lanes sharing each row (xor stays within half-wave)
  #pragma unroll
  for (int d = 1; d < 16; d <<= 1)
    #pragma unroll
    for (int i = 0; i < 8; ++i) {
      float ob = __shfl_xor(best[i], d, 32);
      int   oi = __shfl_xor(bidx[i], d, 32);
      if (ob > best[i] || (ob == best[i] && oi < bidx[i])) { best[i] = ob; bidx[i] = oi; }
    }

  if (ln == 0) {
    #pragma unroll
    for (int i = 0; i < 8; ++i) {
      int m = tile * 16 + i + (g << 3);
      if (m < M) idxOut[m] = bidx[i];
    }
  }
}

// ---------------------------------------------------------------------------
// gather codebook rows -> NCHW h0
// ---------------------------------------------------------------------------
__global__ void gather_kernel(const float* __restrict__ cb, const int* __restrict__ idx,
                              float* __restrict__ h, int pn) {
  int total = NBATCH * CDIM * pn * pn;
  int tid = blockIdx.x * blockDim.x + threadIdx.x;
  if (tid >= total) return;
  int x = tid % pn; int t = tid / pn;
  int y = t % pn;   t /= pn;
  int c = t & 63;   int b = t >> 6;
  int m = (b * pn + y) * pn + x;
  h[tid] = cb[idx[m] * CDIM + c];
}

// ---------------------------------------------------------------------------
// loss = 2 * mean((f_hat - f)^2)
// ---------------------------------------------------------------------------
__global__ void loss_kernel(const float* __restrict__ fhat, const float* __restrict__ f,
                            float* __restrict__ loss) {
  __shared__ float sm[256];
  float s = 0.0f;
  for (int i = blockIdx.x * blockDim.x + threadIdx.x; i < NFELEM; i += gridDim.x * blockDim.x) {
    float d = fhat[i] - f[i];
    s += d * d;
  }
  sm[threadIdx.x] = s;
  __syncthreads();
  for (int st = 128; st > 0; st >>= 1) {
    if (threadIdx.x < st) sm[threadIdx.x] += sm[threadIdx.x + st];
    __syncthreads();
  }
  if (threadIdx.x == 0) atomicAdd(loss, sm[0] * (2.0f / (float)NFELEM));
}

// ---------------------------------------------------------------------------
// host orchestration
// ---------------------------------------------------------------------------
struct CDesc { int k, ostride, pad, ldil, IH, OH; };

static void launch_conv(const float* in, const float* w, const float* bias, float* out,
                        const CDesc& d, int mode, const float* hin, float* fh, float* fr,
                        hipStream_t stream) {
  int M = NBATCH * d.OH * d.OH;
  int Mtiles = (M + 15) / 16;
  int blocks = (Mtiles + 7) / 8;  // 8 waves / block (wave32)
#define CONV_CASE(KK, OSv, PDv, LDv)                                                   \
  if (d.k == KK && d.ostride == OSv && d.pad == PDv && d.ldil == LDv) {                \
    conv_wmma_kernel<KK, OSv, PDv, LDv><<<blocks, 256, 0, stream>>>(                   \
        in, w, bias, out, d.IH, d.OH, mode, hin, fh, fr);                              \
    return;                                                                            \
  }
  CONV_CASE(3, 2, 1, 1)   // stride-2 3x3 (down)
  CONV_CASE(2, 1, 0, 1)   // 2x2 valid (down, pn=1 tail)
  CONV_CASE(5, 2, 0, 1)   // 5x5 s2 (down, pn=6)
  CONV_CASE(7, 1, 0, 1)   // 7x7 valid (down, pn=10)
  CONV_CASE(4, 1, 0, 1)   // 4x4 valid (down, pn=13)
  CONV_CASE(3, 1, 1, 1)   // 3x3 same (down pn=16, up pn=16, Phi)
  CONV_CASE(2, 1, 1, 1)   // convT 2x2 s1 p0 (up, pn=1 head)
  CONV_CASE(4, 1, 2, 2)   // convT 4x4 s2 p1 (up)
  CONV_CASE(6, 1, 5, 2)   // convT 6x6 s2 p0 (up, pn=6)
  CONV_CASE(7, 1, 6, 1)   // convT 7x7 s1 p0 (up, pn=10)
  CONV_CASE(4, 1, 3, 1)   // convT 4x4 s1 p0 (up, pn=13)
#undef CONV_CASE
}

extern "C" void kernel_launch(void* const* d_in, const int* in_sizes, int n_in,
                              void* d_out, int out_size, void* d_ws, size_t ws_size,
                              hipStream_t stream) {
  const float* f  = (const float*)d_in[0];
  const float* cb = (const float*)d_in[1];

  // pytree leaf order: dict keys sorted alphabetically.
  // down/up keys "1","10","13","16","2","4","6","8" -> scale index (PNS order 1,2,4,6,8,10,13,16)
  static const int keyToSi[8] = {0, 5, 6, 7, 1, 2, 3, 4};
  static const int nStages[8] = {4, 3, 2, 1, 1, 1, 1, 1};   // per si, same for down & up
  static const int PNS[8]     = {1, 2, 4, 6, 8, 10, 13, 16};

  const float *dW[8][4], *dB[8][4], *uW[8][4], *uB[8][4], *pW[4], *pB[4];
  int p = 2;
  for (int kk = 0; kk < 8; ++kk) {
    int si = keyToSi[kk];
    for (int s = 0; s < nStages[si]; ++s) {
      dW[si][s] = (const float*)d_in[p++];
      dB[si][s] = (const float*)d_in[p++];
    }
  }
  for (int i = 0; i < 4; ++i) { pW[i] = (const float*)d_in[p++]; pB[i] = (const float*)d_in[p++]; }
  for (int kk = 0; kk < 8; ++kk) {
    int si = keyToSi[kk];
    for (int s = 0; s < nStages[si]; ++s) {
      uW[si][s] = (const float*)d_in[p++];
      uB[si][s] = (const float*)d_in[p++];
    }
  }

  // conv descriptors  (transposed: ostride=1, pad=k-1-p, ldil=s)
  static const CDesc dDesc[8][4] = {
    { {3,2,1,1,16,8}, {3,2,1,1,8,4}, {3,2,1,1,4,2}, {2,1,0,1,2,1} },   // pn=1
    { {3,2,1,1,16,8}, {3,2,1,1,8,4}, {3,2,1,1,4,2}, {0,0,0,0,0,0} },   // pn=2
    { {3,2,1,1,16,8}, {3,2,1,1,8,4}, {0,0,0,0,0,0}, {0,0,0,0,0,0} },   // pn=4
    { {5,2,0,1,16,6}, {0,0,0,0,0,0}, {0,0,0,0,0,0}, {0,0,0,0,0,0} },   // pn=6
    { {3,2,1,1,16,8}, {0,0,0,0,0,0}, {0,0,0,0,0,0}, {0,0,0,0,0,0} },   // pn=8
    { {7,1,0,1,16,10},{0,0,0,0,0,0}, {0,0,0,0,0,0}, {0,0,0,0,0,0} },   // pn=10
    { {4,1,0,1,16,13},{0,0,0,0,0,0}, {0,0,0,0,0,0}, {0,0,0,0,0,0} },   // pn=13
    { {3,1,1,1,16,16},{0,0,0,0,0,0}, {0,0,0,0,0,0}, {0,0,0,0,0,0} },   // pn=16
  };
  static const CDesc uDesc[8][4] = {
    { {2,1,1,1,1,2},  {4,1,2,2,2,4}, {4,1,2,2,4,8}, {4,1,2,2,8,16} },  // pn=1
    { {4,1,2,2,2,4},  {4,1,2,2,4,8}, {4,1,2,2,8,16},{0,0,0,0,0,0} },   // pn=2
    { {4,1,2,2,4,8},  {4,1,2,2,8,16},{0,0,0,0,0,0}, {0,0,0,0,0,0} },   // pn=4
    { {6,1,5,2,6,16}, {0,0,0,0,0,0}, {0,0,0,0,0,0}, {0,0,0,0,0,0} },   // pn=6
    { {4,1,2,2,8,16}, {0,0,0,0,0,0}, {0,0,0,0,0,0}, {0,0,0,0,0,0} },   // pn=8
    { {7,1,6,1,10,16},{0,0,0,0,0,0}, {0,0,0,0,0,0}, {0,0,0,0,0,0} },   // pn=10
    { {4,1,3,1,13,16},{0,0,0,0,0,0}, {0,0,0,0,0,0}, {0,0,0,0,0,0} },   // pn=13
    { {3,1,1,1,16,16},{0,0,0,0,0,0}, {0,0,0,0,0,0}, {0,0,0,0,0,0} },   // pn=16 (normal conv)
  };
  static const CDesc phiDesc = {3, 1, 1, 1, 16, 16};

  // workspace
  float* frest = (float*)d_ws;
  float* bufA  = frest + NFELEM;
  float* bufB  = bufA + NFELEM;
  float* wsq   = bufB + NFELEM;
  int*   idxb  = (int*)(wsq + VOCAB);
  float* fhat  = (float*)d_out;
  float* dloss = fhat + NFELEM;

  vq_init_kernel<<<(NFELEM + 1 + 255) / 256, 256, 0, stream>>>(f, frest, fhat);
  vq_wsq_kernel<<<VOCAB / 256, 256, 0, stream>>>(cb, wsq);

  for (int si = 0; si < 8; ++si) {
    int pn = PNS[si];
    // --- downsample stack: f_rest -> z [96,64,pn,pn]
    const float* cur = frest;
    float* nxt = bufA;
    for (int s = 0; s < nStages[si]; ++s) {
      launch_conv(cur, dW[si][s], dB[si][s], nxt, dDesc[si][s], 0, nullptr, nullptr, nullptr, stream);
      cur = nxt;
      nxt = (cur == bufA) ? bufB : bufA;
    }
    // --- nearest-codebook search (WMMA score GEMM + shuffle argmin)
    int M = NBATCH * pn * pn;
    int Mtiles = (M + 15) / 16;
    argmin_wmma_kernel<<<(Mtiles + 7) / 8, 256, 0, stream>>>(cur, cb, wsq, idxb, pn);
    // --- gather h0 = W[idx] in NCHW
    float* h = nxt;
    gather_kernel<<<(M * CDIM + 255) / 256, 256, 0, stream>>>(cb, idxb, h, pn);
    cur = h;
    nxt = (cur == bufA) ? bufB : bufA;
    // --- upsample stack: h0 -> h [96,64,16,16]
    for (int s = 0; s < nStages[si]; ++s) {
      launch_conv(cur, uW[si][s], uB[si][s], nxt, uDesc[si][s], 0, nullptr, nullptr, nullptr, stream);
      cur = nxt;
      nxt = (cur == bufA) ? bufB : bufA;
    }
    // --- Phi residual mix + accumulate: f_hat += v ; f_rest -= v
    launch_conv(cur, pW[si / 2], pB[si / 2], nullptr, phiDesc, 1, cur, fhat, frest, stream);
  }

  loss_kernel<<<256, 256, 0, stream>>>(fhat, f, dloss);
}
